// LRUCell_4054449128211
// MI455X (gfx1250) — matrix operand order repaired and checked by reading, not compiled, not run on hardware
//
#include <hip/hip_runtime.h>
#include <hip/hip_bf16.h>

#define BATCH 65536
#define HID   256
#define DIN   256

typedef __attribute__((ext_vector_type(16))) __bf16 v16bf;
typedef __attribute__((ext_vector_type(8)))  float  v8f;

union FragAB { v16bf v; unsigned u[8]; };

__device__ __forceinline__ unsigned pack_bf16(float lo, float hi) {
    unsigned a = __builtin_bit_cast(unsigned, lo);
    unsigned b = __builtin_bit_cast(unsigned, hi);
    a = (a + 0x7fffu + ((a >> 16) & 1u)) >> 16;
    b = (b + 0x7fffu + ((b >> 16) & 1u)) >> 16;
    return (a & 0xffffu) | (b << 16);
}

__device__ __forceinline__ unsigned short f2bf(float x) {
    unsigned a = __builtin_bit_cast(unsigned, x);
    return (unsigned short)((a + 0x7fffu + ((a >> 16) & 1u)) >> 16);
}

__device__ __forceinline__ v8f wmma_bf16(const FragAB& a, const FragAB& b, v8f c) {
    return __builtin_amdgcn_wmma_f32_16x16x32_bf16(
        /*neg_a=*/false, a.v, /*neg_b=*/false, b.v,
        /*c_mod=*/(short)0, c, /*reuse_a=*/false, /*reuse_b=*/false);
}

// ---------------------------------------------------------------------------
// Prep: pack weights into WMMA B-fragment layout (bf16) + compute Lambda.
// Packed layout per 32(K)x16(N) fragment: 256 dwords, dword d = lane*8 + v,
// holding bf16 pair W[k0][n], W[k0+1][n], k0 = kb*32 + (v>>2)*16 + (lane>>4)*8 + 2(v&3),
// n = nb*16 + (lane&15). Matrices: 0=B_real*gamma, 1=B_img*gamma, 2=C_real,
// 3=-C_img (negated so pass-B is pure accumulation).
// ---------------------------------------------------------------------------
__global__ __launch_bounds__(256) void lru_prep(
    const float* __restrict__ nu_log, const float* __restrict__ theta_log,
    const float* __restrict__ gamma_log,
    const float* __restrict__ B_real, const float* __restrict__ B_img,
    const float* __restrict__ C_real, const float* __restrict__ C_img,
    unsigned* __restrict__ wsPacked, float* __restrict__ Lr, float* __restrict__ Li)
{
    int p = blockIdx.x * 256 + threadIdx.x;
    if (p < 4 * 32768) {
        int mat  = p >> 15;
        int idx  = p & 32767;
        int frag = idx >> 8;        // (kb*16 + nb)
        int d    = idx & 255;
        int ln   = d >> 3;
        int v    = d & 7;
        int kb   = frag >> 4;
        int nb   = frag & 15;
        int n    = nb * 16 + (ln & 15);
        int k0   = kb * 32 + (v >> 2) * 16 + (ln >> 4) * 8 + 2 * (v & 3);
        const float* src = (mat == 0) ? B_real : (mat == 1) ? B_img
                         : (mat == 2) ? C_real : C_img;
        float s0 = src[k0 * 256 + n];
        float s1 = src[(k0 + 1) * 256 + n];
        if (mat < 2) { float g = expf(gamma_log[n]); s0 *= g; s1 *= g; }
        else if (mat == 3) { s0 = -s0; s1 = -s1; }
        wsPacked[p] = pack_bf16(s0, s1);
    } else if (p < 4 * 32768 + 256) {
        int h = p - 4 * 32768;
        float mag = expf(-expf(nu_log[h]));
        float th  = expf(theta_log[h]);
        Lr[h] = mag * cosf(th);
        Li[h] = mag * sinf(th);
    }
}

// ---------------------------------------------------------------------------
// Main fused kernel: 32 batch rows per block, 8 waves (2 M-tiles x 4 N-slices).
// ---------------------------------------------------------------------------
__global__ __launch_bounds__(256) void lru_main(
    const float* __restrict__ xr_in, const float* __restrict__ xi_in,
    const float* __restrict__ u_in,  const float* __restrict__ Dvec,
    const unsigned* __restrict__ wBgr, const unsigned* __restrict__ wBgi,
    const unsigned* __restrict__ wCr,  const unsigned* __restrict__ wCni,
    const float* __restrict__ Lr, const float* __restrict__ Li,
    float* __restrict__ xr_out, float* __restrict__ xi_out,
    float* __restrict__ y_out)
{
    __shared__ unsigned       sU[32 * 128];    // u tile, bf16 pairs row-major [32][128]
    __shared__ unsigned short sXr[32 * 256];   // x_k real, bf16
    __shared__ unsigned short sXi[32 * 256];   // x_k imag, bf16

    const int tid     = threadIdx.x;
    const int lane    = tid & 31;
    const int w       = tid >> 5;     // wave 0..7
    const int mw      = w >> 2;       // M-tile 0..1
    const int nw      = w & 3;        // N-slice 0..3
    const int rowBase = blockIdx.x * 32;
    const int m16     = lane & 15;
    const int half    = lane >> 4;

    // ---- Stage 1: stage u tile (32x256 f32) into LDS as bf16 pairs ----
    {
        const float2* up = reinterpret_cast<const float2*>(u_in + (size_t)rowBase * 256);
        #pragma unroll
        for (int i = 0; i < 16; ++i) {
            int p = tid + i * 256;           // 4096 pairs
            float2 f = up[p];
            sU[p] = pack_bf16(f.x, f.y);
        }
    }
    __syncthreads();

    // ---- Stage 2: uB_r / uB_i GEMMs (K = DIN = 256 -> 8 k-steps) ----
    v8f zero = {0.f, 0.f, 0.f, 0.f, 0.f, 0.f, 0.f, 0.f};
    v8f accR[4], accI[4];
    #pragma unroll
    for (int t = 0; t < 4; ++t) { accR[t] = zero; accI[t] = zero; }

    #pragma unroll
    for (int kb = 0; kb < 8; ++kb) {
        FragAB a;
        const unsigned* rp = &sU[(mw * 16 + m16) * 128 + kb * 16 + half * 4];
        a.u[0] = rp[0]; a.u[1] = rp[1]; a.u[2] = rp[2];  a.u[3] = rp[3];
        a.u[4] = rp[8]; a.u[5] = rp[9]; a.u[6] = rp[10]; a.u[7] = rp[11];
        #pragma unroll
        for (int t = 0; t < 4; ++t) {
            int nb = nw * 4 + t;
            int fb = (kb * 16 + nb) * 256 + lane * 8;
            FragAB br, bi;
            uint4 lo = reinterpret_cast<const uint4*>(wBgr + fb)[0];
            uint4 hi = reinterpret_cast<const uint4*>(wBgr + fb)[1];
            br.u[0] = lo.x; br.u[1] = lo.y; br.u[2] = lo.z; br.u[3] = lo.w;
            br.u[4] = hi.x; br.u[5] = hi.y; br.u[6] = hi.z; br.u[7] = hi.w;
            lo = reinterpret_cast<const uint4*>(wBgi + fb)[0];
            hi = reinterpret_cast<const uint4*>(wBgi + fb)[1];
            bi.u[0] = lo.x; bi.u[1] = lo.y; bi.u[2] = lo.z; bi.u[3] = lo.w;
            bi.u[4] = hi.x; bi.u[5] = hi.y; bi.u[6] = hi.z; bi.u[7] = hi.w;
            accR[t] = wmma_bf16(a, br, accR[t]);
            accI[t] = wmma_bf16(a, bi, accI[t]);
        }
    }

    // ---- Stage 2 epilogue: x_k = Lambda*x + uB ; write f32 out + bf16 LDS ----
    #pragma unroll
    for (int t = 0; t < 4; ++t) {
        int c  = nw * 64 + t * 16 + m16;
        float lr = Lr[c], li = Li[c];
        #pragma unroll
        for (int v = 0; v < 8; ++v) {
            int rloc = mw * 16 + half * 8 + v;
            size_t g = (size_t)(rowBase + rloc) * 256 + c;
            float xr = xr_in[g], xi = xi_in[g];
            float nr = lr * xr - li * xi + accR[t][v];
            float ni = lr * xi + li * xr + accI[t][v];
            xr_out[g] = nr;
            xi_out[g] = ni;
            sXr[rloc * 256 + c] = f2bf(nr);
            sXi[rloc * 256 + c] = f2bf(ni);
        }
    }
    __syncthreads();

    // ---- Stage 3: y = xr_k @ C_r + xi_k @ (-C_i)  (K = HID = 256) ----
    v8f accY[4];
    #pragma unroll
    for (int t = 0; t < 4; ++t) accY[t] = zero;

    #pragma unroll
    for (int kb = 0; kb < 8; ++kb) {
        FragAB ar, ai;
        {
            const unsigned* rr = reinterpret_cast<const unsigned*>(sXr)
                               + (mw * 16 + m16) * 128 + kb * 16 + half * 4;
            ar.u[0] = rr[0]; ar.u[1] = rr[1]; ar.u[2] = rr[2];  ar.u[3] = rr[3];
            ar.u[4] = rr[8]; ar.u[5] = rr[9]; ar.u[6] = rr[10]; ar.u[7] = rr[11];
            const unsigned* ri = reinterpret_cast<const unsigned*>(sXi)
                               + (mw * 16 + m16) * 128 + kb * 16 + half * 4;
            ai.u[0] = ri[0]; ai.u[1] = ri[1]; ai.u[2] = ri[2];  ai.u[3] = ri[3];
            ai.u[4] = ri[8]; ai.u[5] = ri[9]; ai.u[6] = ri[10]; ai.u[7] = ri[11];
        }
        #pragma unroll
        for (int t = 0; t < 4; ++t) {
            int nb = nw * 4 + t;
            int fb = (kb * 16 + nb) * 256 + lane * 8;
            FragAB bcr, bci;
            uint4 lo = reinterpret_cast<const uint4*>(wCr + fb)[0];
            uint4 hi = reinterpret_cast<const uint4*>(wCr + fb)[1];
            bcr.u[0] = lo.x; bcr.u[1] = lo.y; bcr.u[2] = lo.z; bcr.u[3] = lo.w;
            bcr.u[4] = hi.x; bcr.u[5] = hi.y; bcr.u[6] = hi.z; bcr.u[7] = hi.w;
            lo = reinterpret_cast<const uint4*>(wCni + fb)[0];
            hi = reinterpret_cast<const uint4*>(wCni + fb)[1];
            bci.u[0] = lo.x; bci.u[1] = lo.y; bci.u[2] = lo.z; bci.u[3] = lo.w;
            bci.u[4] = hi.x; bci.u[5] = hi.y; bci.u[6] = hi.z; bci.u[7] = hi.w;
            accY[t] = wmma_bf16(ar, bcr, accY[t]);
            accY[t] = wmma_bf16(ai, bci, accY[t]);
        }
    }

    // ---- Stage 3 epilogue: y += D * u ----
    #pragma unroll
    for (int t = 0; t < 4; ++t) {
        int c  = nw * 64 + t * 16 + m16;
        float dv = Dvec[c];
        #pragma unroll
        for (int v = 0; v < 8; ++v) {
            int rloc = mw * 16 + half * 8 + v;
            size_t g = (size_t)(rowBase + rloc) * 256 + c;
            y_out[g] = accY[t][v] + dv * u_in[g];
        }
    }
}

extern "C" void kernel_launch(void* const* d_in, const int* in_sizes, int n_in,
                              void* d_out, int out_size, void* d_ws, size_t ws_size,
                              hipStream_t stream) {
    const float* x_real    = (const float*)d_in[0];
    const float* x_imag    = (const float*)d_in[1];
    const float* u_k       = (const float*)d_in[2];
    const float* nu_log    = (const float*)d_in[3];
    const float* theta_log = (const float*)d_in[4];
    const float* gamma_log = (const float*)d_in[5];
    const float* B_real    = (const float*)d_in[6];
    const float* B_img     = (const float*)d_in[7];
    const float* C_real    = (const float*)d_in[8];
    const float* C_img     = (const float*)d_in[9];
    const float* Dvec      = (const float*)d_in[10];

    unsigned* wsP = (unsigned*)d_ws;          // 4 * 32768 dwords = 512 KB packed weights
    float* Lr = (float*)(wsP + 4 * 32768);
    float* Li = Lr + 256;

    float* out    = (float*)d_out;
    float* xr_out = out;
    float* xi_out = out + (size_t)BATCH * HID;
    float* y_out  = out + 2 * (size_t)BATCH * HID;

    lru_prep<<<513, 256, 0, stream>>>(nu_log, theta_log, gamma_log,
                                      B_real, B_img, C_real, C_img,
                                      wsP, Lr, Li);

    lru_main<<<BATCH / 32, 256, 0, stream>>>(
        x_real, x_imag, u_k, Dvec,
        wsP, wsP + 32768, wsP + 65536, wsP + 98304,
        Lr, Li, xr_out, xi_out, y_out);
}
